// MoCo_28424093565170
// MI455X (gfx1250) — matrix operand order repaired and checked by reading, not compiled, not run on hardware
//
#include <hip/hip_runtime.h>
#include <hip/hip_bf16.h>

// ---------------------------------------------------------------------------
// MoCo forward on gfx1250 (MI455X).
// Split-bf16 (bf16x2) WMMA GEMMs: A*B ~= Ah*Bh + Ah*Bl + Al*Bh  (f32-class
// accuracy, 3 v_wmma per 16x16 tile per k-step), TDM (tensor_load_to_lds)
// double-buffered staging of f32 queue tiles, fused retrieval argmax.
// BATCH=512, K=65536, DIM=768.
// ---------------------------------------------------------------------------

typedef __attribute__((ext_vector_type(16))) __bf16       v16bf;
typedef __attribute__((ext_vector_type(2)))  __bf16       v2bf;
typedef __attribute__((ext_vector_type(8)))  float        v8f;
typedef __attribute__((ext_vector_type(4)))  unsigned int u32x4;
typedef __attribute__((ext_vector_type(4)))  int          i32x4;
typedef __attribute__((ext_vector_type(8)))  int          i32x8;
typedef __attribute__((ext_vector_type(4)))  float        f32x4;

#define BATCH  512
#define KQ     65536
#define DIM    768
#define DIM2   1536
#define QROWS  64            // queue rows staged per block
#define STAGEK 64            // K f32 per TDM stage (2 WMMA k-steps)
#define KSTEP  32            // K per WMMA
#define LDSROW 68            // padded f32 row stride in LDS (64 + 4 pad)
#define NCHUNK (KQ / QROWS)  // 1024

#if defined(__gfx1250__) && __has_builtin(__builtin_amdgcn_tensor_load_to_lds)
#define HAS_TDM 1
#else
#define HAS_TDM 0
#endif

// f32 -> bf16 RNE (scalar)
__device__ __forceinline__ unsigned short f2bf(float f) {
    unsigned u = __float_as_uint(f);
    return (unsigned short)((u + 0x7FFFu + ((u >> 16) & 1u)) >> 16);
}

// packed f32x2 -> bf16x2 (v_cvt_pk_bf16_f32 when available)
__device__ __forceinline__ unsigned cvt2(float x, float y) {
#if defined(__gfx1250__) && __has_builtin(__builtin_amdgcn_cvt_pk_bf16_f32)
    union { v2bf b; unsigned u; } r;
    r.b = __builtin_amdgcn_cvt_pk_bf16_f32(x, y);
    return r.u;
#else
    return (unsigned)f2bf(x) | ((unsigned)f2bf(y) << 16);
#endif
}

// exact split x = hi + lo (hi,lo bf16); x-hi is exact in f32 (Sterbenz)
__device__ __forceinline__ void split_bf(float x, unsigned short* hi, unsigned short* lo) {
    unsigned short h = f2bf(x);
    float hf_ = __uint_as_float((unsigned)h << 16);
    *hi = h;
    *lo = f2bf(x - hf_);
}

// A-fragment: 16x32 bf16 row-major, 16 contiguous K per lane (k = 16*hf + s).
__device__ __forceinline__ v16bf load_a_frag(const unsigned short* rowp, int k0, int hf) {
    const unsigned short* p = rowp + k0 + hf * 16;
    union { u32x4 u[2]; v16bf v; } r;
    r.u[0] = *(const u32x4*)(p);
    r.u[1] = *(const u32x4*)(p + 8);
    return r.v;
}

// B-fragment from bf16 [N,K]: 16 contiguous K per lane.
__device__ __forceinline__ v16bf load_b_bf16(const unsigned short* p) {
    union { u32x4 u[2]; v16bf v; } r;
    r.u[0] = *(const u32x4*)(p);
    r.u[1] = *(const u32x4*)(p + 8);
    return r.v;
}

struct bpair { v16bf h, l; };

// 16 contiguous f32 (LDS) -> split bf16 hi/lo fragments, in-register.
__device__ __forceinline__ bpair load_b_split(const float* p) {
    union { unsigned u[8]; v16bf v; } H, L;
#pragma unroll
    for (int i = 0; i < 4; i++) {
        f32x4 f = *(const f32x4*)(p + 4 * i);
        unsigned h0 = cvt2(f.x, f.y);
        unsigned h1 = cvt2(f.z, f.w);
        float hx = __uint_as_float(h0 << 16);
        float hy = __uint_as_float(h0 & 0xFFFF0000u);
        float hz = __uint_as_float(h1 << 16);
        float hw = __uint_as_float(h1 & 0xFFFF0000u);
        H.u[2 * i]     = h0;
        H.u[2 * i + 1] = h1;
        L.u[2 * i]     = cvt2(f.x - hx, f.y - hy);
        L.u[2 * i + 1] = cvt2(f.z - hz, f.w - hw);
    }
    bpair r;
    r.h = H.v;
    r.l = L.v;
    return r;
}

__device__ __forceinline__ v8f wmma_bf16(v16bf a, v16bf b, v8f c) {
    return __builtin_amdgcn_wmma_f32_16x16x32_bf16(false, a, false, b, (short)0, c,
                                                   false, false);
}

// split product accumulate: acc += Ah*Bh + Ah*Bl + Al*Bh
__device__ __forceinline__ v8f wmma_split(v16bf ah, v16bf al, bpair b, v8f acc) {
    acc = wmma_bf16(al, b.h, acc);
    acc = wmma_bf16(ah, b.l, acc);
    acc = wmma_bf16(ah, b.h, acc);
    return acc;
}

#if HAS_TDM
// TDM: 2D tile (STAGEK cols x QROWS rows, f32) queue -> LDS, 4-DWORD pad per
// 64-DWORD row (perfect 64-bank spread for b128 LDS reads). D# per ISA ch.8.
__device__ __forceinline__ void tdm_issue(const float* gsrc, void* lds_dst) {
    unsigned long long ga = (unsigned long long)(uintptr_t)gsrc;
    unsigned lds = (unsigned)(uintptr_t)lds_dst;   // low 32 bits = LDS offset
    u32x4 g0;
    g0[0] = 1u;                                               // count=1
    g0[1] = lds;                                              // lds_addr
    g0[2] = (unsigned)ga;                                     // global_addr lo
    g0[3] = (unsigned)((ga >> 32) & 0x1FFFFFFu) | (2u << 30); // ga hi | type=2
    i32x8 g1;
    g1[0] = (2 << 16)      // data_size = 4B
          | (1 << 20)      // pad_enable
          | (5 << 22)      // pad_interval: 64 DWORDs
          | (3 << 25);     // pad_amount: 4 DWORDs
    g1[1] = (int)((DIM & 0xFFFFu) << 16);                 // tensor_dim0[15:0]
    g1[2] = (int)((DIM >> 16) | ((KQ & 0xFFFF) << 16));   // td0 hi | td1 lo
    g1[3] = (int)((KQ >> 16) | (STAGEK << 16));           // td1 hi | tile_dim0
    g1[4] = (int)QROWS;                                   // tile_dim1, tile_dim2=0
    g1[5] = DIM;                                          // tensor_dim0_stride
    g1[6] = 0;
    g1[7] = 0;
    i32x4 z4 = {};
#if __clang_major__ >= 23
    i32x8 z8 = {};
    __builtin_amdgcn_tensor_load_to_lds(g0, g1, z4, z4, z8, 0);
#else
    __builtin_amdgcn_tensor_load_to_lds(g0, g1, z4, z4, 0);
#endif
}
#endif

// ---------------------------------------------------------------------------
// K0: L2-normalize rows; f32 -> new-queue rows [0,512), split bf16 into the
// first DIM cols of Xc (img) / Xd (txt). grid (512,2), 256 thr.
// ---------------------------------------------------------------------------
__global__ void k_normalize(const float* __restrict__ img, const float* __restrict__ txt,
                            unsigned short* __restrict__ Xc_h, unsigned short* __restrict__ Xc_l,
                            unsigned short* __restrict__ Xd_h, unsigned short* __restrict__ Xd_l,
                            float* __restrict__ nq_img, float* __restrict__ nq_txt) {
    int row = blockIdx.x;
    const float*    src  = (blockIdx.y == 0) ? img : txt;
    unsigned short* dh   = (blockIdx.y == 0) ? Xc_h : Xd_h;
    unsigned short* dl   = (blockIdx.y == 0) ? Xc_l : Xd_l;
    float*          dstq = (blockIdx.y == 0) ? nq_img : nq_txt;

    __shared__ float red[256];
    float s = 0.f;
    for (int c = threadIdx.x; c < DIM; c += 256) {
        float v = src[row * DIM + c];
        s += v * v;
    }
    red[threadIdx.x] = s;
    __syncthreads();
    for (int off = 128; off > 0; off >>= 1) {
        if (threadIdx.x < off) red[threadIdx.x] += red[threadIdx.x + off];
        __syncthreads();
    }
    float inv = 1.0f / fmaxf(sqrtf(red[0]), 1e-12f);
    for (int c = threadIdx.x; c < DIM; c += 256) {
        float v = src[row * DIM + c] * inv;
        dstq[row * DIM + c] = v;
        unsigned short h, l;
        split_bf(v, &h, &l);
        dh[row * DIM2 + c] = h;
        dl[row * DIM2 + c] = l;
    }
}

// ---------------------------------------------------------------------------
// K1: W [DIM2, DIM] f32 -> Wt_hi/Wt_lo [DIM, DIM2] bf16 (transpose + split).
// ---------------------------------------------------------------------------
__global__ void k_cvt_w(const float* __restrict__ W,
                        unsigned short* __restrict__ Wh, unsigned short* __restrict__ Wl) {
    int idx = blockIdx.x * 256 + threadIdx.x;
    int n = idx / DIM2;
    int k = idx - n * DIM2;
    unsigned short h, l;
    split_bf(W[k * DIM + n], &h, &l);
    Wh[idx] = h;
    Wl[idx] = l;
}

// ---------------------------------------------------------------------------
// Big queue GEMM, MODE 0: retrieval + fused argmax partials.
//               MODE 1: logits = scale * A @ Q^T, full store.
// Block: 256 thr = 8 waves = 4 m-tiles x 2 n-halves; tile M=64, N=64.
// f32 queue tile staged in LDS via TDM (double buffered, STAGEK=64);
// split-bf16 B fragments derived in-register. grid (KQ/64, BATCH/64).
// ---------------------------------------------------------------------------
template <int MODE>
__global__ void k_qgemm(const unsigned short* __restrict__ Ah,
                        const unsigned short* __restrict__ Al, int lda,
                        const float* __restrict__ Q,
                        float* __restrict__ pval, int* __restrict__ pidx,
                        const float* __restrict__ scale_p, float* __restrict__ out) {
    __shared__ float qb[2][QROWS * LDSROW];
    __shared__ float sv[8][16];
    __shared__ int   si[8][16];

    int chunk = blockIdx.x;
    int mblk  = blockIdx.y * 64;
    int tid   = threadIdx.x;
    int wave  = tid >> 5, lane = tid & 31;
    int ll = lane & 15, hf = lane >> 4;
    int mt = wave >> 1, nh = wave & 1;
    int mbase = mblk + mt * 16;
    long long qbase = (long long)chunk * QROWS;

    const unsigned short* aph = Ah + (mbase + ll) * lda;
    const unsigned short* apl = Al + (mbase + ll) * lda;
    int r0 = nh * 32 + ll;    // LDS row (queue row within tile) of acc0 column
    int r1 = r0 + 16;

    v8f acc0 = {}, acc1 = {};

#if HAS_TDM
    if (wave == 0) {
        tdm_issue(Q + qbase * DIM, &qb[0][0]);
        __builtin_amdgcn_s_wait_tensorcnt(0);
    }
    __syncthreads();
    int cur = 0;
    for (int st = 0; st < DIM / STAGEK; st++) {
        if (wave == 0 && st + 1 < DIM / STAGEK)
            tdm_issue(Q + qbase * DIM + (st + 1) * STAGEK, &qb[cur ^ 1][0]);
#pragma unroll
        for (int sub = 0; sub < 2; sub++) {
            int k0 = st * STAGEK + sub * KSTEP;
            v16bf ah = load_a_frag(aph, k0, hf);
            v16bf al = load_a_frag(apl, k0, hf);
            const float* bp = &qb[cur][0] + sub * KSTEP + hf * 16;
            bpair b0 = load_b_split(bp + r0 * LDSROW);
            bpair b1 = load_b_split(bp + r1 * LDSROW);
            acc0 = wmma_split(ah, al, b0, acc0);
            acc1 = wmma_split(ah, al, b1, acc1);
        }
        if (wave == 0) __builtin_amdgcn_s_wait_tensorcnt(0);
        __syncthreads();          // next buffer ready; this buffer fully read
        cur ^= 1;
    }
#else
    for (int st = 0; st < DIM / STAGEK; st++) {
        // cooperative staging: 64 rows x 16 float4 = 1024 float4, 4 per thread
#pragma unroll
        for (int it = 0; it < 4; it++) {
            int idx = tid + it * 256;
            int rr = idx >> 4, qq = idx & 15;
            *(f32x4*)&qb[0][rr * LDSROW + qq * 4] =
                *(const f32x4*)(Q + (qbase + rr) * DIM + st * STAGEK + qq * 4);
        }
        __syncthreads();
#pragma unroll
        for (int sub = 0; sub < 2; sub++) {
            int k0 = st * STAGEK + sub * KSTEP;
            v16bf ah = load_a_frag(aph, k0, hf);
            v16bf al = load_a_frag(apl, k0, hf);
            const float* bp = &qb[0][0] + sub * KSTEP + hf * 16;
            bpair b0 = load_b_split(bp + r0 * LDSROW);
            bpair b1 = load_b_split(bp + r1 * LDSROW);
            acc0 = wmma_split(ah, al, b0, acc0);
            acc1 = wmma_split(ah, al, b1, acc1);
        }
        __syncthreads();
    }
#endif

    if (MODE == 1) {
        float s = scale_p[0];
        long long qg0 = qbase + r0, qg1 = qbase + r1;
#pragma unroll
        for (int v = 0; v < 8; v++) {
            long long r = mbase + hf * 8 + v;
            out[r * (long long)KQ + qg0] = s * acc0[v];
            out[r * (long long)KQ + qg1] = s * acc1[v];
        }
        return;
    }

    // MODE 0: per-row argmax over this block's 64 columns
    int qg0 = (int)(qbase + r0), qg1 = (int)(qbase + r1);
    float bv[8];
    int   bi[8];
#pragma unroll
    for (int v = 0; v < 8; v++) {
        bool t0 = acc1[v] > acc0[v];
        float mv = t0 ? acc1[v] : acc0[v];
        int   mi = t0 ? qg1 : qg0;
#pragma unroll
        for (int off = 1; off < 16; off <<= 1) {
            float ov = __shfl_xor(mv, off, 32);
            int   oi = __shfl_xor(mi, off, 32);
            bool tk = (ov > mv) || (ov == mv && oi < mi);
            mv = tk ? ov : mv;
            mi = tk ? oi : mi;
        }
        bv[v] = mv;
        bi[v] = mi;
    }
    if (ll == 0) {
#pragma unroll
        for (int v = 0; v < 8; v++) {
            sv[wave][hf * 8 + v] = bv[v];
            si[wave][hf * 8 + v] = bi[v];
        }
    }
    __syncthreads();
    if (tid < 64) {
        int lr = tid & 15, m2 = tid >> 4;
        float v0 = sv[m2 * 2][lr], v1 = sv[m2 * 2 + 1][lr];
        int   i0 = si[m2 * 2][lr], i1 = si[m2 * 2 + 1][lr];
        bool tk = (v1 > v0) || (v1 == v0 && i1 < i0);
        float mv = tk ? v1 : v0;
        int   mi = tk ? i1 : i0;
        int r = mblk + tid;
        pval[(long long)r * NCHUNK + chunk] = mv;
        pidx[(long long)r * NCHUNK + chunk] = mi;
    }
}

// ---------------------------------------------------------------------------
// K3: reduce NCHUNK partials per row -> ind_similar.
// ---------------------------------------------------------------------------
__global__ void k_argmax_fin(const float* __restrict__ pval, const int* __restrict__ pidx,
                             int* __restrict__ ind, float* __restrict__ out_ind) {
    int row = blockIdx.x, t = threadIdx.x;
    __shared__ float sv[256];
    __shared__ int   si[256];
    float mv = -3.402823466e38f;
    int   mi = 0x7fffffff;
    for (int c = t; c < NCHUNK; c += 256) {
        float v = pval[(long long)row * NCHUNK + c];
        int   i = pidx[(long long)row * NCHUNK + c];
        bool tk = (v > mv) || (v == mv && i < mi);
        mv = tk ? v : mv;
        mi = tk ? i : mi;
    }
    sv[t] = mv;
    si[t] = mi;
    __syncthreads();
    for (int off = 128; off > 0; off >>= 1) {
        if (t < off) {
            bool tk = (sv[t + off] > sv[t]) || (sv[t + off] == sv[t] && si[t + off] < si[t]);
            if (tk) { sv[t] = sv[t + off]; si[t] = si[t + off]; }
        }
        __syncthreads();
    }
    if (t == 0) {
        ind[row]     = si[0];
        out_ind[row] = (float)si[0];
    }
}

// ---------------------------------------------------------------------------
// K4: gather t_similar = queue_txt[ind[n]] -> split bf16 into Xd[:, DIM:2*DIM].
// ---------------------------------------------------------------------------
__global__ void k_gather(const float* __restrict__ Qtxt, const int* __restrict__ ind,
                         unsigned short* __restrict__ Xd_h, unsigned short* __restrict__ Xd_l) {
    int n = blockIdx.x;
    int src = ind[n];
    for (int c = threadIdx.x; c < DIM; c += 256) {
        unsigned short h, l;
        split_bf(Qtxt[(long long)src * DIM + c], &h, &l);
        Xd_h[n * DIM2 + DIM + c] = h;
        Xd_l[n * DIM2 + DIM + c] = l;
    }
}

// ---------------------------------------------------------------------------
// K5: small GEMM  D = act(A[BATCH,Kd] @ Wt^T + bias), split in / split out.
// Wt is [N,Kd]. grid (N/64, BATCH/16), 128 thr.
// ---------------------------------------------------------------------------
__global__ void k_gemm_small(const unsigned short* __restrict__ Ah,
                             const unsigned short* __restrict__ Al, int lda,
                             const unsigned short* __restrict__ Bh,
                             const unsigned short* __restrict__ Bl, int Kd,
                             const float* __restrict__ bias, int relu,
                             unsigned short* __restrict__ Dh,
                             unsigned short* __restrict__ Dl, int ldd, int dcol0) {
    int tid = threadIdx.x;
    int wave = tid >> 5, lane = tid & 31;
    int ll = lane & 15, hf = lane >> 4;
    int mbase = blockIdx.y * 16;
    int nbase = blockIdx.x * 64 + wave * 16;

    const unsigned short* aph = Ah + (mbase + ll) * lda;
    const unsigned short* apl = Al + (mbase + ll) * lda;
    const unsigned short* bph = Bh + (nbase + ll) * Kd + hf * 16;
    const unsigned short* bpl = Bl + (nbase + ll) * Kd + hf * 16;

    v8f acc = {};
    for (int k0 = 0; k0 < Kd; k0 += KSTEP) {
        v16bf ah = load_a_frag(aph, k0, hf);
        v16bf al = load_a_frag(apl, k0, hf);
        bpair b;
        b.h = load_b_bf16(bph + k0);
        b.l = load_b_bf16(bpl + k0);
        acc = wmma_split(ah, al, b, acc);
    }
#pragma unroll
    for (int v = 0; v < 8; v++) {
        int r = mbase + hf * 8 + v;
        int c = nbase + ll;
        float x = acc[v] + bias[c];
        if (relu) x = fmaxf(x, 0.f);
        unsigned short h, l;
        split_bf(x, &h, &l);
        Dh[r * ldd + dcol0 + c] = h;
        Dl[r * ldd + dcol0 + c] = l;
    }
}

// ---------------------------------------------------------------------------
// Host launcher
// ---------------------------------------------------------------------------
extern "C" void kernel_launch(void* const* d_in, const int* in_sizes, int n_in,
                              void* d_out, int out_size, void* d_ws, size_t ws_size,
                              hipStream_t stream) {
    const float* img  = (const float*)d_in[0];
    const float* txt  = (const float*)d_in[1];
    const float* qimg = (const float*)d_in[2];
    const float* qtxt = (const float*)d_in[3];
    const float* Wd   = (const float*)d_in[4];
    const float* bd   = (const float*)d_in[5];
    const float* Wc   = (const float*)d_in[6];
    const float* bc   = (const float*)d_in[7];
    const float* ls   = (const float*)d_in[8];

    float* out_logits = (float*)d_out;
    float* out_ind    = out_logits + (size_t)BATCH * KQ;
    float* out_qimg   = out_ind + BATCH;
    float* out_qtxt   = out_qimg + (size_t)KQ * DIM;

    char* w = (char*)d_ws;
    unsigned short* Xd_h = (unsigned short*)(w + 0);          // [512,1536]
    unsigned short* Xd_l = (unsigned short*)(w + 1572864);
    unsigned short* Xc_h = (unsigned short*)(w + 3145728);
    unsigned short* Xc_l = (unsigned short*)(w + 4718592);
    unsigned short* Wd_h = (unsigned short*)(w + 6291456);    // [768,1536]
    unsigned short* Wd_l = (unsigned short*)(w + 8650752);
    unsigned short* Wc_h = (unsigned short*)(w + 11010048);
    unsigned short* Wc_l = (unsigned short*)(w + 13369344);
    unsigned short* cb_h = (unsigned short*)(w + 15728640);   // [512,768]
    unsigned short* cb_l = (unsigned short*)(w + 16515072);
    float*          pval = (float*)(w + 17301504);            // [512,1024]
    int*            pidx = (int*)(w + 19398656);
    int*            ind  = (int*)(w + 21495808);              // [512]

    hipMemcpyAsync(out_qimg, qimg, (size_t)KQ * DIM * sizeof(float),
                   hipMemcpyDeviceToDevice, stream);
    hipMemcpyAsync(out_qtxt, qtxt, (size_t)KQ * DIM * sizeof(float),
                   hipMemcpyDeviceToDevice, stream);

    k_normalize<<<dim3(BATCH, 2), 256, 0, stream>>>(img, txt, Xc_h, Xc_l, Xd_h, Xd_l,
                                                    out_qimg, out_qtxt);
    k_cvt_w<<<(DIM * DIM2) / 256, 256, 0, stream>>>(Wd, Wd_h, Wd_l);
    k_cvt_w<<<(DIM * DIM2) / 256, 256, 0, stream>>>(Wc, Wc_h, Wc_l);

    // retrieval GEMM + fused argmax (split-bf16, ~f32 accurate)
    k_qgemm<0><<<dim3(NCHUNK, BATCH / 64), 256, 0, stream>>>(
        Xc_h, Xc_l, DIM2, qtxt, pval, pidx, nullptr, nullptr);
    k_argmax_fin<<<BATCH, 256, 0, stream>>>(pval, pidx, ind, out_ind);
    k_gather<<<BATCH, 256, 0, stream>>>(qtxt, ind, Xd_h, Xd_l);

    // diff = relu([txt_f | t_similar] @ W_diff + b_diff) -> Xc[:, 768:]
    k_gemm_small<<<dim3(DIM / 64, BATCH / 16), 128, 0, stream>>>(
        Xd_h, Xd_l, DIM2, Wd_h, Wd_l, DIM2, bd, 1, Xc_h, Xc_l, DIM2, DIM);
    // combined = [img_f | diff] @ W_comb + b_comb -> comb (split bf16)
    k_gemm_small<<<dim3(DIM / 64, BATCH / 16), 128, 0, stream>>>(
        Xc_h, Xc_l, DIM2, Wc_h, Wc_l, DIM2, bc, 0, cb_h, cb_l, DIM, 0);

    // logits = scale * combined @ queue_img^T
    k_qgemm<1><<<dim3(NCHUNK, BATCH / 64), 256, 0, stream>>>(
        cb_h, cb_l, DIM, qimg, nullptr, nullptr, ls, out_logits);
}
